// CausalSelfAttention_90417651516063
// MI455X (gfx1250) — compile-verified, matrix operand order checked by reading, not run on hardware
//
#include <hip/hip_runtime.h>

// ---------------------------------------------------------------- types
typedef _Float16 v16h __attribute__((ext_vector_type(16)));
typedef _Float16 h8   __attribute__((ext_vector_type(8)));
typedef _Float16 h4   __attribute__((ext_vector_type(4)));
typedef float    v8f  __attribute__((ext_vector_type(8)));

// ---------------------------------------------------------------- constants
constexpr int Bc   = 2;
constexpr int Tc   = 4096;
constexpr int Cc   = 1024;
constexpr int Hc   = 8;
constexpr int KVHc = 2;
constexpr int Dc   = 128;
constexpr int REP  = Hc / KVHc;          // 4
constexpr int QKVC = Cc + 2 * KVHc * Dc; // 1536
constexpr float SCALE = 0.08838834764831845f; // 1/sqrt(128)

__device__ inline v8f wmma16(v16h a, v16h b, v8f c) {
  // v_wmma_f32_16x16x32_f16  (neg_a, A, neg_b, B, c_mod, C, reuse_a, reuse_b)
  return __builtin_amdgcn_wmma_f32_16x16x32_f16(false, a, false, b, (short)0, c,
                                                false, false);
}

// Build a 16-element operand fragment from two contiguous 8xf16 (16B) chunks.
// 16-bit operand layout (16x32): lane<16 -> elems 0..7 = K 0..7, 8..15 = K 16..23
//                                lane>=16 -> K 8..15 and K 24..31
__device__ inline v16h load_frag(const _Float16* base0, const _Float16* base1) {
  h8 lo = *(const h8*)base0;
  h8 hi = *(const h8*)base1;
  v16h r;
#pragma unroll
  for (int i = 0; i < 8; ++i) { r[i] = lo[i]; r[i + 8] = hi[i]; }
  return r;
}

// Async global->LDS copy of 16 bytes (GLOBAL_LOAD_ASYNC_TO_LDS_B128, GV mode).
// LDS destination = low 32 bits of the generic shared-pointer (LDS aperture:
// addr[31:0] is the LDS byte offset). Tracked by ASYNCcnt, in-order completion.
__device__ inline void async_copy_b128(_Float16* lds_dst, const _Float16* gsrc) {
  unsigned dst = (unsigned)(unsigned long long)(void*)lds_dst;
  asm volatile("global_load_async_to_lds_b128 %0, %1, off"
               :
               : "v"(dst), "v"(gsrc)
               : "memory");
}
template <int N>
__device__ inline void wait_async() {
  asm volatile("s_wait_asynccnt %0" ::"i"(N) : "memory");
}

// ---------------------------------------------------------------- kernel 1: prep
// x -> f16; Wq|Wk|Wv packed TRANSPOSED [1536][1024]; Wproj TRANSPOSED [1024][1024]
__global__ void prep_kernel(const float* __restrict__ x,
                            const float* __restrict__ Wq,
                            const float* __restrict__ Wk,
                            const float* __restrict__ Wv,
                            const float* __restrict__ Wproj,
                            _Float16* __restrict__ xh,
                            _Float16* __restrict__ wqkvT,
                            _Float16* __restrict__ wprojT) {
  size_t i = (size_t)blockIdx.x * blockDim.x + threadIdx.x;
  size_t stride = (size_t)gridDim.x * blockDim.x;
  const size_t NX = (size_t)Bc * Tc * Cc;
  for (size_t j = i; j < NX; j += stride) xh[j] = (_Float16)x[j];
  const size_t NQKV = (size_t)QKVC * Cc;
  for (size_t j = i; j < NQKV; j += stride) {
    int c = (int)(j / Cc), k = (int)(j % Cc);   // c = output col, k = input dim
    float v;
    if (c < Cc)            v = Wq[(size_t)k * Cc + c];
    else if (c < Cc + 256) v = Wk[(size_t)k * 256 + (c - Cc)];
    else                   v = Wv[(size_t)k * 256 + (c - Cc - 256)];
    wqkvT[j] = (_Float16)v;
  }
  const size_t NP = (size_t)Cc * Cc;
  for (size_t j = i; j < NP; j += stride) {
    int n = (int)(j / Cc), k = (int)(j % Cc);
    wprojT[j] = (_Float16)Wproj[(size_t)k * Cc + n];
  }
}

// ---------------------------------------------------------------- kernel 2/5: WMMA GEMM
// D[M x N] = A[M x K] (f16 row-major) * BmT[N x K] (f16 row-major = B^T)
// CTA: 256 threads = 8 waves in 2(M) x 4(N); CTA tile 64 x 128; wave tile 32x32.
// Double-buffered GLOBAL_LOAD_ASYNC_TO_LDS_B128 staging (3 async ops / wave /
// stage; in-order completion => s_wait_asynccnt 3 retires the older stage).
template <typename OutT>
__global__ __launch_bounds__(256) void wmma_gemm(const _Float16* __restrict__ A,
                                                 const _Float16* __restrict__ BmT,
                                                 OutT* __restrict__ D,
                                                 int M, int N, int K) {
  __shared__ alignas(16) _Float16 lA[2][64][40];   // 32 K + 8 pad (80B rows)
  __shared__ alignas(16) _Float16 lB[2][128][40];  // B^T tile: [n][k]

  const int tid  = threadIdx.x;
  const int lane = tid & 31;
  const int wv   = tid >> 5;
  const int wm   = wv >> 2;          // 0..1
  const int wn   = wv & 3;           // 0..3
  const int g    = lane >> 4;        // half-wave
  const int ln   = lane & 15;

  const int m0 = blockIdx.y * 64;
  const int n0 = blockIdx.x * 128;

  v8f acc[2][2] = {};

  const int la_m  = tid >> 2;        // 0..63, chunk = tid&3 (8 f16 each)
  const int la_ch = tid & 3;
  const int lb_n  = tid >> 1;        // 0..127, half = tid&1 (16 f16 each)
  const int lb_h  = tid & 1;

  const _Float16* srcA0 = A   + (size_t)(m0 + la_m) * K + la_ch * 8;
  const _Float16* srcB0 = BmT + (size_t)(n0 + lb_n) * K + lb_h * 16;

  auto stage = [&](int kk, int buf) {
    async_copy_b128(&lA[buf][la_m][la_ch * 8],     srcA0 + kk);
    async_copy_b128(&lB[buf][lb_n][lb_h * 16],     srcB0 + kk);
    async_copy_b128(&lB[buf][lb_n][lb_h * 16 + 8], srcB0 + kk + 8);
  };

  stage(0, 0);
  for (int kk = 0; kk < K; kk += 32) {
    const int cur = (kk >> 5) & 1;
    const bool more = (kk + 32 < K);
    if (more) stage(kk + 32, cur ^ 1);     // overlap next DMA with this compute
    if (more) wait_async<3>();             // older stage done, newer in flight
    else      wait_async<0>();
    __syncthreads();

    v16h af[2], bf[2];
#pragma unroll
    for (int mt = 0; mt < 2; ++mt) {
      const int row = wm * 32 + mt * 16 + ln;
      af[mt] = load_frag(&lA[cur][row][8 * g], &lA[cur][row][16 + 8 * g]);
    }
#pragma unroll
    for (int nt = 0; nt < 2; ++nt) {
      const int col = wn * 32 + nt * 16 + ln;
      bf[nt] = load_frag(&lB[cur][col][8 * g], &lB[cur][col][16 + 8 * g]);
    }
#pragma unroll
    for (int mt = 0; mt < 2; ++mt)
#pragma unroll
      for (int nt = 0; nt < 2; ++nt)
        acc[mt][nt] = wmma16(af[mt], bf[nt], acc[mt][nt]);
    __syncthreads();   // readers done before buffer is re-staged
  }

  // epilogue: C/D layout -> lane ln = col, row = r + 8*g within 16x16 tile
#pragma unroll
  for (int mt = 0; mt < 2; ++mt)
#pragma unroll
    for (int nt = 0; nt < 2; ++nt)
#pragma unroll
      for (int r = 0; r < 8; ++r) {
        const int row = m0 + wm * 32 + mt * 16 + r + 8 * g;
        const int col = n0 + wn * 32 + nt * 16 + ln;
        D[(size_t)row * N + col] = (OutT)acc[mt][nt][r];
      }
}

// ---------------------------------------------------------------- kernel 3: QKV epilogue
// One wave per (b,t,head-row). q/k: rotary+rms -> head-major [b][h][t][d].
// v: gate+VE -> TRANSPOSED [b][kvh][d][t] so attention PV needs no transpose.
__global__ void qkv_epilogue(const _Float16* __restrict__ qkv_raw,
                             const float* __restrict__ x,
                             const float* __restrict__ ve,
                             const float* __restrict__ cosb,
                             const float* __restrict__ sinb,
                             const float* __restrict__ Wg,
                             _Float16* __restrict__ qh,
                             _Float16* __restrict__ kh,
                             _Float16* __restrict__ vT) {
  const int NQW = Bc * Tc * Hc;
  const int NKW = Bc * Tc * KVHc;
  const int w    = blockIdx.x * (blockDim.x >> 5) + (threadIdx.x >> 5);
  const int lane = threadIdx.x & 31;
  const int d0   = lane * 4;

  float v[4];

  if (w < NQW + NKW) {
    // ---- q or k: rotary then rms_norm
    int h, bt, colbase;
    _Float16* dstbase;
    if (w < NQW) {
      h = w % Hc; bt = w / Hc; colbase = h * Dc;
      dstbase = qh;
    } else {
      int w2 = w - NQW;
      h = w2 % KVHc; bt = w2 / KVHc; colbase = Cc + h * Dc;
      dstbase = kh;
    }
    const int b = bt >> 12, t = bt & (Tc - 1);
    const _Float16* src = qkv_raw + (size_t)bt * QKVC + colbase;
#pragma unroll
    for (int i = 0; i < 4; ++i) v[i] = (float)src[d0 + i];
    // rotary: pair (d, d+64); partner lives in lane^16 at same sub-index
#pragma unroll
    for (int i = 0; i < 4; ++i) {
      float other = __shfl_xor(v[i], 16, 32);
      int dd = d0 + i;
      if (dd < 64) {
        float c = cosb[t * 64 + dd], s = sinb[t * 64 + dd];
        v[i] = v[i] * c + other * s;
      } else {
        int d2 = dd - 64;
        float c = cosb[t * 64 + d2], s = sinb[t * 64 + d2];
        v[i] = -other * s + v[i] * c;
      }
    }
    float ss = v[0] * v[0] + v[1] * v[1] + v[2] * v[2] + v[3] * v[3];
#pragma unroll
    for (int m = 1; m <= 16; m <<= 1) ss += __shfl_xor(ss, m, 32);
    float r = rsqrtf(ss * (1.0f / 128.0f) + 1.1920929e-7f);
    h4 o;
#pragma unroll
    for (int i = 0; i < 4; ++i) o[i] = (_Float16)(v[i] * r);
    size_t dst;
    if (w < NQW) dst = (((size_t)(b * Hc + h) * Tc + t) * Dc) + d0;
    else         dst = (((size_t)(b * KVHc + h) * Tc + t) * Dc) + d0;
    *(h4*)(dstbase + dst) = o;
  } else if (w < NQW + 2 * NKW) {
    // ---- v: add gate * ve, store transposed [b][kvh][d][t]
    int w3 = w - NQW - NKW;
    int h = w3 % KVHc, bt = w3 / KVHc;
    const int b = bt >> 12, t = bt & (Tc - 1);
    const _Float16* src = qkv_raw + (size_t)bt * QKVC + Cc + 256 + h * Dc;
#pragma unroll
    for (int i = 0; i < 4; ++i) v[i] = (float)src[d0 + i];
    float p = x[(size_t)bt * Cc + lane] * Wg[lane * KVHc + h];
#pragma unroll
    for (int m = 1; m <= 16; m <<= 1) p += __shfl_xor(p, m, 32);
    float gate = 2.0f / (1.0f + __expf(-p));
    const float* vesrc = ve + (size_t)bt * (KVHc * Dc) + h * Dc + d0;
    _Float16* base = vT + ((size_t)(b * KVHc + h) * Dc) * Tc + t;
#pragma unroll
    for (int i = 0; i < 4; ++i)
      base[(size_t)(d0 + i) * Tc] = (_Float16)(v[i] + gate * vesrc[i]);
  }
}

// ---------------------------------------------------------------- kernel 4: flash attention
// Block = 8 waves (256 threads); each wave owns 16 queries, steps 32 keys.
// V is pre-transposed, so PV B-fragments are contiguous global loads (L2-hot:
// V is 4 MB vs 192 MB L2). Only ldsP (P relayout) remains in LDS.
__global__ __launch_bounds__(256, 1) void attn_kernel(
    const _Float16* __restrict__ qh,
    const _Float16* __restrict__ kh,
    const _Float16* __restrict__ vT,
    _Float16* __restrict__ yh,
    const int* __restrict__ winp) {
  __shared__ alignas(16) _Float16 ldsP[8][16][40];    // probs 16x32 per wave

  const int lane = threadIdx.x & 31;
  const int wv   = threadIdx.x >> 5;
  const int g    = lane >> 4;
  const int ln   = lane & 15;

  const int tiles = Tc / 128;
  const int bid = blockIdx.x;
  const int b    = bid / (Hc * tiles);
  const int rem  = bid % (Hc * tiles);
  const int h    = rem / tiles;
  const int tile = rem % tiles;
  const int q0   = tile * 128 + wv * 16;
  const int kvh  = h / REP;
  const int win  = *winp;

  const _Float16* qbase  = qh + ((size_t)(b * Hc + h) * Tc) * Dc;
  const _Float16* kbase  = kh + ((size_t)(b * KVHc + kvh) * Tc) * Dc;
  const _Float16* vTbase = vT + ((size_t)(b * KVHc + kvh) * Dc) * Tc;

  // Q fragments (reused for every key step): 4 d-chunks of 32
  v16h aq[4];
  {
    const _Float16* qrow = qbase + (size_t)(q0 + ln) * Dc;
#pragma unroll
    for (int dc = 0; dc < 4; ++dc)
      aq[dc] = load_frag(qrow + dc * 32 + 8 * g, qrow + dc * 32 + 16 + 8 * g);
  }

  float m[8], s[8];
#pragma unroll
  for (int r = 0; r < 8; ++r) { m[r] = -1e30f; s[r] = 0.0f; }
  v8f acc[8] = {};   // 16 rows x 128 d as 8 C-tiles

  int kstart = q0 - win;
  if (kstart < 0) kstart = 0;
  kstart &= ~31;

  for (int kb = kstart; kb <= q0 + 15; kb += 32) {
    // ---- scores: 2 N-tiles x 4 d-chunks (K rows contiguous -> b128 global)
    v8f sc[2] = {};
#pragma unroll
    for (int nt = 0; nt < 2; ++nt) {
      int key = kb + nt * 16 + ln;
      if (key >= Tc) key = Tc - 1;  // masked below anyway
      const _Float16* krow = kbase + (size_t)key * Dc;
#pragma unroll
      for (int dc = 0; dc < 4; ++dc) {
        v16h bk = load_frag(krow + dc * 32 + 8 * g, krow + dc * 32 + 16 + 8 * g);
        sc[nt] = wmma16(aq[dc], bk, sc[nt]);
      }
    }

    // Interior key-tiles need no masking (wave-uniform test).
    const bool needmask = (kb + 31 > q0) || (kb < q0 + 15 - win);

    // ---- online softmax per row (r, half); reductions in 16-lane halves
#pragma unroll
    for (int r = 0; r < 8; ++r) {
      float e0, e1;
      if (needmask) {
        const int row = q0 + r + 8 * g;
        const int c0 = kb + ln, c1 = kb + 16 + ln;
        e0 = (c0 > row || c0 < row - win) ? -1e30f : sc[0][r] * SCALE;
        e1 = (c1 > row || c1 < row - win) ? -1e30f : sc[1][r] * SCALE;
      } else {
        e0 = sc[0][r] * SCALE;
        e1 = sc[1][r] * SCALE;
      }
      float mx = fmaxf(e0, e1);
#pragma unroll
      for (int msk = 1; msk <= 8; msk <<= 1) mx = fmaxf(mx, __shfl_xor(mx, msk, 32));
      float mn = fmaxf(m[r], mx);
      float f  = __expf(m[r] - mn);
      float p0 = __expf(e0 - mn);
      float p1 = __expf(e1 - mn);
      float rs = p0 + p1;
#pragma unroll
      for (int msk = 1; msk <= 8; msk <<= 1) rs += __shfl_xor(rs, msk, 32);
      s[r] = s[r] * f + rs;
      m[r] = mn;
#pragma unroll
      for (int dt = 0; dt < 8; ++dt) acc[dt][r] *= f;   // rescale in place
      ldsP[wv][r + 8 * g][ln]      = (_Float16)p0;
      ldsP[wv][r + 8 * g][16 + ln] = (_Float16)p1;
    }

    // ---- PV: P (16x32) as A; V^T rows give B columns straight from global
    v16h ap = load_frag(&ldsP[wv][ln][8 * g], &ldsP[wv][ln][16 + 8 * g]);
#pragma unroll
    for (int dt = 0; dt < 8; ++dt) {
      const _Float16* vrow = vTbase + (size_t)(dt * 16 + ln) * Tc + kb;
      v16h bv = load_frag(vrow + 8 * g, vrow + 16 + 8 * g);
      acc[dt] = wmma16(ap, bv, acc[dt]);
    }
  }

  // ---- normalize and write y[b, t, h*128 + d] (f16 for the output GEMM)
#pragma unroll
  for (int dt = 0; dt < 8; ++dt)
#pragma unroll
    for (int r = 0; r < 8; ++r) {
      const int row = q0 + r + 8 * g;
      float val = acc[dt][r] / s[r];
      yh[((size_t)(b * Tc + row)) * Cc + h * Dc + dt * 16 + ln] = (_Float16)val;
    }
}

// ---------------------------------------------------------------- launch
extern "C" void kernel_launch(void* const* d_in, const int* in_sizes, int n_in,
                              void* d_out, int out_size, void* d_ws, size_t ws_size,
                              hipStream_t stream) {
  const float* x     = (const float*)d_in[0];
  const float* ve    = (const float*)d_in[1];
  const float* cosb  = (const float*)d_in[2];
  const float* sinb  = (const float*)d_in[3];
  const float* Wq    = (const float*)d_in[4];
  const float* Wk    = (const float*)d_in[5];
  const float* Wv    = (const float*)d_in[6];
  const float* Wproj = (const float*)d_in[7];
  const float* Wg    = (const float*)d_in[8];
  const int*   winp  = (const int*)d_in[9];

  char* ws = (char*)d_ws;
  size_t off = 0;
  auto alloc = [&](size_t bytes) -> void* {
    void* p = ws + off;
    off = (off + bytes + 255) & ~(size_t)255;
    return p;
  };
  const size_t BT = (size_t)Bc * Tc;               // 8192
  _Float16* xh      = (_Float16*)alloc(BT * Cc * 2);
  _Float16* wqkvT   = (_Float16*)alloc((size_t)QKVC * Cc * 2);
  _Float16* wprojT  = (_Float16*)alloc((size_t)Cc * Cc * 2);
  _Float16* qkv_raw = (_Float16*)alloc(BT * QKVC * 2);
  _Float16* qh      = (_Float16*)alloc(BT * Hc * Dc * 2);
  _Float16* kh      = (_Float16*)alloc(BT * KVHc * Dc * 2);
  _Float16* vT      = (_Float16*)alloc(BT * KVHc * Dc * 2 + 256); // +pad: tail reads
  _Float16* yh      = (_Float16*)alloc(BT * Cc * 2);

  // 1) convert / pack (weights stored transposed for contiguous B tiles)
  prep_kernel<<<2048, 256, 0, stream>>>(x, Wq, Wk, Wv, Wproj, xh, wqkvT, wprojT);

  // 2) fused QKV GEMM: [8192 x 1024] @ [1024 x 1536]
  {
    dim3 grid(QKVC / 128, (int)(BT / 64));
    wmma_gemm<_Float16><<<grid, 256, 0, stream>>>(xh, wqkvT, qkv_raw,
                                                  (int)BT, QKVC, Cc);
  }

  // 3) rotary + rmsnorm + VE gate (V written transposed)
  {
    int waves = Bc * Tc * Hc + 2 * Bc * Tc * KVHc;  // 98304
    qkv_epilogue<<<waves / 8, 256, 0, stream>>>(qkv_raw, x, ve, cosb, sinb, Wg,
                                                qh, kh, vT);
  }

  // 4) windowed flash attention
  attn_kernel<<<Bc * Hc * (Tc / 128), 256, 0, stream>>>(qh, kh, vT, yh, winp);

  // 5) output projection straight into d_out (fp32)
  {
    dim3 grid(Cc / 128, (int)(BT / 64));
    wmma_gemm<float><<<grid, 256, 0, stream>>>(yh, wprojT, (float*)d_out,
                                               (int)BT, Cc, Cc);
  }
}
